// MultiHeadAttention_7713761264360
// MI455X (gfx1250) — matrix operand order repaired and checked by reading, not compiled, not run on hardware
//
#include <hip/hip_runtime.h>
#include <hip/hip_bf16.h>

// ---------------------------------------------------------------------------
// Types for CDNA5 WMMA (wave32): bf16 16x16x32, f32 accumulate
// ---------------------------------------------------------------------------
typedef __bf16 bf16_t;
typedef __attribute__((ext_vector_type(16))) __bf16 v16bf;
typedef __attribute__((ext_vector_type(8)))  float  v8f;
typedef __attribute__((ext_vector_type(4)))  unsigned int v4u;   // POD 16B chunk

struct Pair16 { v4u a, b; };   // 32 bytes, bit-castable to v16bf

static __device__ inline v16bf load_frag(const bf16_t* lo, const bf16_t* hi) {
  Pair16 p;
  p.a = *(const v4u*)lo;   // 16B aligned
  p.b = *(const v4u*)hi;   // 16B aligned
  return __builtin_bit_cast(v16bf, p);
}

static __device__ inline v8f wmma_bf16(v16bf a, v16bf b, v8f c) {
  return __builtin_amdgcn_wmma_f32_16x16x32_bf16(false, a, false, b, (short)0, c,
                                                 false, false);
}

// gfx1250 async global->LDS copy (ASYNCcnt tracked), 16B per lane.
// lds_off: LDS byte address (low 32 bits of flat shared pointer);
// gaddr:   64-bit global address.
static __device__ inline void async_g2l_b128(unsigned lds_off,
                                             unsigned long long gaddr) {
  asm volatile("global_load_async_to_lds_b128 %0, %1, off"
               :: "v"(lds_off), "v"(gaddr) : "memory");
}
static __device__ inline void wait_asynccnt0() {
  asm volatile("s_wait_asynccnt 0x0" ::: "memory");
}

// Problem constants
#define Bq     8
#define SEGq   512
#define MEMq   512
#define TOTq   1024
#define MODELq 1024
#define Hq     16
#define Dq     64

// ---------------------------------------------------------------------------
// Kernel 1: build bf16 copies: h=[mem;x] (8x1024x1024), R (1024x1024),
// and the 5 weight matrices concatenated (wq,wk,wv,wr,wmlp).
// ---------------------------------------------------------------------------
__global__ __launch_bounds__(256) void prep_kernel(
    const float* __restrict__ x, const float* __restrict__ memp,
    const float* __restrict__ R,
    const float* __restrict__ wq, const float* __restrict__ wk,
    const float* __restrict__ wv, const float* __restrict__ wr,
    const float* __restrict__ wm,
    bf16_t* __restrict__ hb, bf16_t* __restrict__ Rb, bf16_t* __restrict__ Wb)
{
  const long N_HB = 8L * 1024 * 1024;
  const long N_RB = 1L * 1024 * 1024;
  const long TOTAL = N_HB + N_RB + 5L * 1024 * 1024;
  for (long idx = (long)blockIdx.x * blockDim.x + threadIdx.x; idx < TOTAL;
       idx += (long)gridDim.x * blockDim.x) {
    if (idx < N_HB) {
      long b = idx >> 20;
      long rem = idx & 0xFFFFF;
      long t = rem >> 10;
      long m = rem & 1023;
      float v = (t < MEMq) ? memp[(b * SEGq + t) * MODELq + m]
                           : x[(b * SEGq + (t - MEMq)) * MODELq + m];
      hb[idx] = (bf16_t)v;
    } else if (idx < N_HB + N_RB) {
      long i = idx - N_HB;
      Rb[i] = (bf16_t)R[i];
    } else {
      long wi = idx - N_HB - N_RB;
      long which = wi >> 20;
      long off = wi & 0xFFFFF;
      const float* src = (which == 0) ? wq
                       : (which == 1) ? wk
                       : (which == 2) ? wv
                       : (which == 3) ? wr : wm;
      Wb[wi] = (bf16_t)src[off];
    }
  }
}

// ---------------------------------------------------------------------------
// Kernel 2: generic tiled bf16 GEMM, C = A[M,1024] x B[1024,1024].
// Block = 256 thr (8 waves); block tile 128(M) x 64(N); wave tile 16x64.
// A logical row r maps to global row (r/group)*gstride + rbase + r%group.
// mode: 0 = f32 out, 1 = bf16 out, 2 = f32 out + residual, 3 = bf16 transposed
//       (idx = (r/tgroup)*tstride + n*tgroup + r%tgroup)
// A tile staged with gfx1250 async global->LDS; B tile transposed in LDS so
// both WMMA operand fragments are contiguous 16B ds_load_b128.
// ---------------------------------------------------------------------------
__global__ __launch_bounds__(256) void gemm_bf16_kernel(
    const bf16_t* __restrict__ A, const bf16_t* __restrict__ B,
    void* __restrict__ Out, const float* __restrict__ resid,
    int M, int group, int gstride, int rbase, int mode,
    int tgroup, long tstride)
{
  __shared__ __align__(16) bf16_t lds_a[128][40];   // 32 cols used, pad to 40
  __shared__ __align__(16) bf16_t lds_bt[64][40];   // B transposed: [n][k]

  const int tid  = threadIdx.x;
  const int wave = tid >> 5;
  const int lane = tid & 31;
  const int l16  = lane & 15;
  const int half = lane >> 4;
  const int m_block = blockIdx.x * 128;
  const int n_block = blockIdx.y * 64;

  // ---- hoisted staging addresses (constant across the K loop) ----
  // A: two 16B chunks per thread
  unsigned           a_lds[2];
  unsigned long long a_glb[2];
  #pragma unroll
  for (int c = 0; c < 2; ++c) {
    int id = tid * 2 + c;          // 0..511
    int row = id >> 2;             // 0..127
    int colc = (id & 3) * 8;       // 0,8,16,24
    int r = m_block + row;
    long gr = (long)(r / group) * gstride + rbase + (r % group);
    a_lds[c] = (unsigned)(unsigned long long)&lds_a[row][colc];
    a_glb[c] = (unsigned long long)(A + gr * 1024 + colc);
  }
  // B: one 16B chunk per thread, scattered transposed
  const int krow = tid >> 3;       // 0..31
  const int nc   = (tid & 7) * 8;  // 0..56
  const bf16_t* bsrc = B + (long)krow * 1024 + n_block + nc;

  v8f acc[4] = {};

  for (int k0 = 0; k0 < 1024; k0 += 32) {
    // ---- stage A tile 128x32 via async global->LDS (ASYNCcnt) ----
    async_g2l_b128(a_lds[0], a_glb[0] + (unsigned long long)k0 * sizeof(bf16_t));
    async_g2l_b128(a_lds[1], a_glb[1] + (unsigned long long)k0 * sizeof(bf16_t));
    // ---- stage B tile 32x64, transposed into LDS ----
    {
      v4u bv = *(const v4u*)(bsrc + (long)k0 * 1024);
      const bf16_t* be = (const bf16_t*)&bv;
      #pragma unroll
      for (int e = 0; e < 8; ++e) lds_bt[nc + e][krow] = be[e];
      if (k0 + 32 < 1024)          // gfx1250 prefetch of next K tile
        __builtin_prefetch(bsrc + (long)(k0 + 32) * 1024, 0, 1);
    }
    wait_asynccnt0();
    __syncthreads();

    // ---- fragments: load everything first, then issue the 4 WMMAs ----
    const int m = wave * 16 + l16;
    v16bf af = load_frag(&lds_a[m][8 * half], &lds_a[m][16 + 8 * half]);
    v16bf b0 = load_frag(&lds_bt[ 0 + l16][8 * half], &lds_bt[ 0 + l16][16 + 8 * half]);
    v16bf b1 = load_frag(&lds_bt[16 + l16][8 * half], &lds_bt[16 + l16][16 + 8 * half]);
    v16bf b2 = load_frag(&lds_bt[32 + l16][8 * half], &lds_bt[32 + l16][16 + 8 * half]);
    v16bf b3 = load_frag(&lds_bt[48 + l16][8 * half], &lds_bt[48 + l16][16 + 8 * half]);
    acc[0] = wmma_bf16(af, b0, acc[0]);
    acc[1] = wmma_bf16(af, b1, acc[1]);
    acc[2] = wmma_bf16(af, b2, acc[2]);
    acc[3] = wmma_bf16(af, b3, acc[3]);
    __syncthreads();
  }

  // ---- epilogue ----
  #pragma unroll
  for (int j = 0; j < 4; ++j) {
    #pragma unroll
    for (int p = 0; p < 8; ++p) {
      int r = m_block + wave * 16 + p + 8 * half;
      int n = n_block + j * 16 + l16;
      float v = acc[j][p];
      if (mode == 0) {
        ((float*)Out)[(long)r * 1024 + n] = v;
      } else if (mode == 1) {
        ((bf16_t*)Out)[(long)r * 1024 + n] = (bf16_t)v;
      } else if (mode == 2) {
        long i = (long)r * 1024 + n;
        ((float*)Out)[i] = v + resid[i];
      } else {
        long i = (long)(r / tgroup) * tstride + (long)n * tgroup + (r % tgroup);
        ((bf16_t*)Out)[i] = (bf16_t)v;
      }
    }
  }
}

// ---------------------------------------------------------------------------
// Kernel 3: fused rel-pos attention.
// Grid: B*H*(SEG/16) blocks; block = 128 thr (4 waves).
// Per block: 16 q rows of one (b,h). Computes ac=(q+u1)k^T and bd=(q+u2)r^T
// via WMMA into LDS, applies circular shift + causal mask + softmax, then
// p @ v via WMMA (v pre-transposed per batch).
// ---------------------------------------------------------------------------
__global__ __launch_bounds__(128) void attn_kernel(
    const float* __restrict__ qf, const bf16_t* __restrict__ kb,
    const bf16_t* __restrict__ vtb, const bf16_t* __restrict__ rb,
    const float* __restrict__ u1, const float* __restrict__ u2,
    const float* __restrict__ amask, bf16_t* __restrict__ attb)
{
  extern __shared__ __align__(16) char smem[];
  float*  s_ac = (float*)smem;                       // 16*1024 f32 (64KB)
  float*  s_bd = (float*)(smem + 65536);             // 16*1024 f32 (64KB)
  bf16_t* a1   = (bf16_t*)(smem + 131072);           // 16*64 bf16 (2KB)
  bf16_t* a2   = (bf16_t*)(smem + 131072 + 2048);    // 16*64 bf16 (2KB)
  float*  red  = (float*)(smem + 131072 + 4096);     // 16*8 f32

  const int tid  = threadIdx.x;
  const int wave = tid >> 5;
  const int lane = tid & 31;
  const int l16  = lane & 15;
  const int half = lane >> 4;
  const int itile = blockIdx.x & 31;
  const int bh = blockIdx.x >> 5;
  const int b  = bh >> 4;
  const int h  = bh & 15;
  const int i0 = itile * 16;

  // ---- phase 0: stage (q+u1), (q+u2) as bf16 (16x64) ----
  {
    int row = tid >> 3;           // 0..15
    int c0 = (tid & 7) * 8;       // 0..56
    const float* qrow = qf + (long)(b * SEGq + i0 + row) * MODELq + h * Dq;
    #pragma unroll
    for (int e = 0; e < 8; ++e) {
      float qv = qrow[c0 + e];
      a1[row * 64 + c0 + e] = (bf16_t)(qv + u1[h * Dq + c0 + e]);
      a2[row * 64 + c0 + e] = (bf16_t)(qv + u2[h * Dq + c0 + e]);
    }
  }
  __syncthreads();

  // ---- phase 1: scores. wave w covers j in [w*256, w*256+256) ----
  const bf16_t* kbase = kb + (long)b * TOTq * MODELq + h * Dq;
  const bf16_t* rbase = rb + h * Dq;
  for (int jt = 0; jt < 16; ++jt) {
    int j0 = wave * 256 + jt * 16;
    v8f acc_ac = {};
    v8f acc_bd = {};
    #pragma unroll
    for (int kc = 0; kc < 64; kc += 32) {
      v16bf af1 = load_frag(&a1[l16 * 64 + kc + 8 * half],
                            &a1[l16 * 64 + kc + 16 + 8 * half]);
      v16bf af2 = load_frag(&a2[l16 * 64 + kc + 8 * half],
                            &a2[l16 * 64 + kc + 16 + 8 * half]);
      const bf16_t* kp = kbase + (long)(j0 + l16) * MODELq + kc + 8 * half;
      const bf16_t* rp = rbase + (long)(j0 + l16) * MODELq + kc + 8 * half;
      v16bf bk = load_frag(kp, kp + 16);
      v16bf br = load_frag(rp, rp + 16);
      acc_ac = wmma_bf16(af1, bk, acc_ac);
      acc_bd = wmma_bf16(af2, br, acc_bd);
    }
    #pragma unroll
    for (int p = 0; p < 8; ++p) {
      int rr = p + 8 * half;
      s_ac[rr * 1024 + j0 + l16] = acc_ac[p];
      s_bd[rr * 1024 + j0 + l16] = acc_bd[p];
    }
  }
  __syncthreads();

  // ---- phase 2: shift + mask + softmax. 8 threads per row ----
  {
    const int row = tid >> 3;
    const int sub = tid & 7;
    const int i_glob = i0 + row;
    float lmax = -__builtin_inff();
    for (int jj = 0; jj < 128; ++jj) {
      int j = sub * 128 + jj;
      int js = (j - i_glob + SEGq - 1) & (TOTq - 1);
      float sv = ((j - i_glob) <= MEMq)
                     ? (s_ac[row * 1024 + j] + s_bd[row * 1024 + js])
                     : 0.0f;
      sv *= 0.125f;                       // 1/sqrt(64)
      if (sv == 0.0f) sv = -__builtin_inff();   // reference quirk
      s_ac[row * 1024 + j] = sv;
      lmax = fmaxf(lmax, sv);
    }
    red[row * 8 + sub] = lmax;
    __syncthreads();
    float rmax = -__builtin_inff();
    for (int s2 = 0; s2 < 8; ++s2) rmax = fmaxf(rmax, red[row * 8 + s2]);
    __syncthreads();
    float lsum = 0.0f;
    for (int jj = 0; jj < 128; ++jj) {
      int j = sub * 128 + jj;
      float p = __expf(s_ac[row * 1024 + j] - rmax);
      s_ac[row * 1024 + j] = p;
      lsum += p;
    }
    red[row * 8 + sub] = lsum;
    __syncthreads();
    float rsum = 0.0f;
    for (int s2 = 0; s2 < 8; ++s2) rsum += red[row * 8 + s2];
    float scale = amask[b * SEGq + i_glob] / rsum;
    for (int jj = 0; jj < 128; ++jj) {
      int j = sub * 128 + jj;
      s_ac[row * 1024 + j] *= scale;
    }
  }
  __syncthreads();

  // ---- phase 3: p @ v. wave w owns dd slice [w*16, w*16+16) ----
  {
    const int dd0 = wave * 16;
    v8f acc = {};
    const bf16_t* vcol = vtb + (long)b * TOTq * MODELq +
                         (long)(h * Dq + dd0 + l16) * TOTq;
    for (int k0 = 0; k0 < 1024; k0 += 32) {
      v16bf af;
      const float* prow = &s_ac[l16 * 1024 + k0];
      #pragma unroll
      for (int e = 0; e < 8; ++e) af[e] = (bf16_t)prow[8 * half + e];
      #pragma unroll
      for (int e = 0; e < 8; ++e) af[8 + e] = (bf16_t)prow[16 + 8 * half + e];
      const bf16_t* vp = vcol + k0 + 8 * half;
      v16bf bfr = load_frag(vp, vp + 16);
      acc = wmma_bf16(af, bfr, acc);
    }
    #pragma unroll
    for (int p = 0; p < 8; ++p) {
      int ii = i0 + p + 8 * half;
      attb[(long)(b * SEGq + ii) * MODELq + h * Dq + dd0 + l16] = (bf16_t)acc[p];
    }
  }
}

// ---------------------------------------------------------------------------
// Kernel 4: row LayerNorm (4096 rows of 1024)
// ---------------------------------------------------------------------------
__global__ __launch_bounds__(256) void ln_kernel(
    const float* __restrict__ y, const float* __restrict__ gamma,
    const float* __restrict__ beta, float* __restrict__ out)
{
  __shared__ float red[256];
  const int r = blockIdx.x;
  const float* yr = y + (long)r * 1024;
  float s = 0.f;
  for (int c = threadIdx.x; c < 1024; c += 256) s += yr[c];
  red[threadIdx.x] = s;
  __syncthreads();
  for (int off = 128; off > 0; off >>= 1) {
    if (threadIdx.x < off) red[threadIdx.x] += red[threadIdx.x + off];
    __syncthreads();
  }
  float mu = red[0] * (1.0f / 1024.0f);
  __syncthreads();
  float v = 0.f;
  for (int c = threadIdx.x; c < 1024; c += 256) {
    float d = yr[c] - mu;
    v += d * d;
  }
  red[threadIdx.x] = v;
  __syncthreads();
  for (int off = 128; off > 0; off >>= 1) {
    if (threadIdx.x < off) red[threadIdx.x] += red[threadIdx.x + off];
    __syncthreads();
  }
  float rstd = rsqrtf(red[0] * (1.0f / 1024.0f) + 1e-5f);
  float* outr = out + (long)r * 1024;
  for (int c = threadIdx.x; c < 1024; c += 256)
    outr[c] = (yr[c] - mu) * rstd * gamma[c] + beta[c];
}

// ---------------------------------------------------------------------------
// Launcher
// ---------------------------------------------------------------------------
extern "C" void kernel_launch(void* const* d_in, const int* in_sizes, int n_in,
                              void* d_out, int out_size, void* d_ws, size_t ws_size,
                              hipStream_t stream)
{
  (void)in_sizes; (void)n_in; (void)out_size; (void)ws_size;
  const float* x     = (const float*)d_in[0];
  const float* memp  = (const float*)d_in[1];
  const float* amask = (const float*)d_in[2];
  const float* wq    = (const float*)d_in[3];
  const float* wk    = (const float*)d_in[4];
  const float* wv    = (const float*)d_in[5];
  const float* wr    = (const float*)d_in[6];
  const float* wm    = (const float*)d_in[7];
  const float* u1    = (const float*)d_in[8];
  const float* u2    = (const float*)d_in[9];
  const float* R     = (const float*)d_in[10];
  const float* gamma = (const float*)d_in[11];
  const float* beta  = (const float*)d_in[12];

  char* ws = (char*)d_ws;
  const size_t MB = 1024ull * 1024ull;
  bf16_t* hb   = (bf16_t*)(ws + 0);        // 16 MB : [8,1024,1024] bf16
  bf16_t* Rb   = (bf16_t*)(ws + 16 * MB);  //  2 MB
  bf16_t* Wb   = (bf16_t*)(ws + 18 * MB);  // 10 MB : wq,wk,wv,wr,wm bf16
  float*  qf   = (float*) (ws + 28 * MB);  // 16 MB : q f32 [4096,1024]
  bf16_t* kb   = (bf16_t*)(ws + 44 * MB);  // 16 MB : k bf16 [8,1024,1024]
  bf16_t* vtb  = (bf16_t*)(ws + 60 * MB);  // 16 MB : v^T bf16 [8,1024(n),1024(t)]
  bf16_t* rb   = (bf16_t*)(ws + 76 * MB);  //  2 MB : r bf16 [1024,1024]
  bf16_t* attb = (bf16_t*)(ws + 78 * MB);  //  8 MB : att bf16 [4096,1024]
  float*  yf   = (float*) (ws + 86 * MB);  // 16 MB : out+x f32

  const size_t ME = 1024ull * 1024ull;     // elements per weight matrix
  bf16_t* wq_b = Wb + 0 * ME;
  bf16_t* wk_b = Wb + 1 * ME;
  bf16_t* wv_b = Wb + 2 * ME;
  bf16_t* wr_b = Wb + 3 * ME;
  bf16_t* wm_b = Wb + 4 * ME;

  prep_kernel<<<4096, 256, 0, stream>>>(x, memp, R, wq, wk, wv, wr, wm, hb, Rb, Wb);

  dim3 blk(256);
  // q = x @ wq (x rows are the second half of each batch's hb block) -> f32
  gemm_bf16_kernel<<<dim3(32, 16), blk, 0, stream>>>(hb, wq_b, qf, nullptr,
      4096, 512, 1024, 512, /*mode*/0, 0, 0);
  // k = h @ wk -> bf16
  gemm_bf16_kernel<<<dim3(64, 16), blk, 0, stream>>>(hb, wk_b, kb, nullptr,
      8192, 8192, 0, 0, /*mode*/1, 0, 0);
  // v = h @ wv -> bf16, transposed per batch: vtb[b][n][t]
  gemm_bf16_kernel<<<dim3(64, 16), blk, 0, stream>>>(hb, wv_b, vtb, nullptr,
      8192, 8192, 0, 0, /*mode*/3, 1024, (long)1024 * 1024);
  // r = R @ wr -> bf16
  gemm_bf16_kernel<<<dim3(8, 16), blk, 0, stream>>>(Rb, wr_b, rb, nullptr,
      1024, 1024, 0, 0, /*mode*/1, 0, 0);

  // fused rel-pos attention (dyn LDS = 2*64KB + 2*2KB + 512B)
  attn_kernel<<<4096, 128, 135680, stream>>>(qf, kb, vtb, rb, u1, u2, amask, attb);

  // out = att @ mlp_w + x -> f32
  gemm_bf16_kernel<<<dim3(32, 16), blk, 0, stream>>>(attb, wm_b, yf, x,
      4096, 4096, 0, 0, /*mode*/2, 0, 0);

  ln_kernel<<<4096, 256, 0, stream>>>(yf, gamma, beta, (float*)d_out);
}
